// LIONBlock_38328288149902
// MI455X (gfx1250) — matrix operand order repaired and checked by reading, not compiled, not run on hardware
//
#include <hip/hip_runtime.h>
#include <math.h>

// ---------------- problem constants ----------------
#define N_TOT   131072          // B*NPB
#define C_DIM   128
#define H_NUM   4
#define DH_DIM  32
#define GS_LEN  4096
#define NGRP    32              // N_TOT / GS
#define T_CH    64
#define NCH     64              // GS / T
#define NB_SORT 512             // N_TOT / 256

typedef __attribute__((ext_vector_type(16))) __bf16 bf16x16;
typedef __attribute__((ext_vector_type(8)))  float  f32x8;

union Frag { bf16x16 v; unsigned int u[8]; uint4 q2[2]; };

__device__ __forceinline__ unsigned short f2bf(float f) {
  unsigned int u = __float_as_uint(f);
  u += 0x7FFFu + ((u >> 16) & 1u);            // round-to-nearest-even
  return (unsigned short)(u >> 16);
}
__device__ __forceinline__ float bf2f(unsigned int hs) {
  return __uint_as_float(hs << 16);
}
__device__ __forceinline__ unsigned int packbf2(float lo, float hi) {
  return (unsigned int)f2bf(lo) | ((unsigned int)f2bf(hi) << 16);
}
__device__ __forceinline__ f32x8 zero8() {
  f32x8 z; for (int i = 0; i < 8; ++i) z[i] = 0.0f; return z;
}
__device__ __forceinline__ f32x8 wmma_bf16(const Frag& a, const Frag& b, f32x8 c) {
  return __builtin_amdgcn_wmma_f32_16x16x32_bf16(false, a.v, false, b.v,
                                                 (short)0, c, false, false);
}

// ---------------- weight packing into B-fragment order ----------------
__global__ __launch_bounds__(256) void pack_w(const float* __restrict__ W,
                                              unsigned* __restrict__ dst, float scale) {
  int frag = blockIdx.x;                 // 0..31
  int lane = threadIdx.x >> 3;           // 0..31
  int j    = threadIdx.x & 7;            // 0..7
  int kt = frag >> 3, nt = frag & 7;
  int n  = nt * 16 + (lane & 15);
  int kk = kt * 32 + ((lane & 16) ? 16 : 0) + 2 * j;
  dst[(frag * 32 + lane) * 8 + j] =
      packbf2(W[kk * C_DIM + n] * scale, W[(kk + 1) * C_DIM + n] * scale);
}

// ---------------- sort keys ----------------
// mx=my=37, mz=2, per=2738, wvol=5408, wy*wz=416, wz=32
__global__ __launch_bounds__(256) void compute_keys(const int* __restrict__ coords,
                                                    unsigned* __restrict__ keys,
                                                    unsigned* __restrict__ idx, int which) {
  int i = blockIdx.x * 256 + threadIdx.x;
  int b = coords[4 * i + 0], z = coords[4 * i + 1];
  int y = coords[4 * i + 2], x = coords[4 * i + 3];
  int wcx = x / 13, wcy = y / 13, wcz = z / 32;
  int cx = x % 13, cy = y % 13, cz = z % 32;
  unsigned key;
  if (which == 0) {
    unsigned bw = (unsigned)(b * 2738 + wcx * 74 + wcy * 2 + wcz);
    key = bw * 5408u + (unsigned)(cx * 416 + cy * 32 + cz);
  } else {
    unsigned bw = (unsigned)(b * 2738 + wcy * 74 + wcx * 2 + wcz);
    key = bw * 5408u + (unsigned)(cy * 416 + cx * 32 + cz);
  }
  keys[i] = key;
  idx[i] = (unsigned)i;
}

__global__ __launch_bounds__(256) void radix_hist(const unsigned* __restrict__ keys,
                                                  unsigned* __restrict__ hist, int shift) {
  __shared__ unsigned h[256];
  h[threadIdx.x] = 0; __syncthreads();
  unsigned d = (keys[blockIdx.x * 256 + threadIdx.x] >> shift) & 255u;
  atomicAdd(&h[d], 1u); __syncthreads();
  hist[threadIdx.x * NB_SORT + blockIdx.x] = h[threadIdx.x];
}

__global__ __launch_bounds__(256) void radix_scan(unsigned* __restrict__ hist) {
  __shared__ unsigned sums[256];
  const int CH = 256 * NB_SORT / 256;     // 512 entries / thread
  int t = threadIdx.x;
  unsigned s = 0;
  for (int j = 0; j < CH; ++j) s += hist[t * CH + j];
  sums[t] = s; __syncthreads();
  if (t == 0) {
    unsigned run = 0;
    for (int i = 0; i < 256; ++i) { unsigned v = sums[i]; sums[i] = run; run += v; }
  }
  __syncthreads();
  unsigned run = sums[t];
  for (int j = 0; j < CH; ++j) { unsigned v = hist[t * CH + j]; hist[t * CH + j] = run; run += v; }
}

__global__ __launch_bounds__(256) void radix_scatter(const unsigned* __restrict__ keys,
                                                     const unsigned* __restrict__ vals,
                                                     unsigned* __restrict__ okeys,
                                                     unsigned* __restrict__ ovals,
                                                     const unsigned* __restrict__ hist, int shift) {
  __shared__ unsigned char dig[256];
  int t = threadIdx.x;
  int gid = blockIdx.x * 256 + t;
  unsigned key = keys[gid], val = vals[gid];
  unsigned d = (key >> shift) & 255u;
  dig[t] = (unsigned char)d; __syncthreads();
  unsigned rank = 0;
  for (int j = 0; j < t; ++j) rank += (dig[j] == (unsigned char)d) ? 1u : 0u;
  unsigned pos = hist[d * NB_SORT + blockIdx.x] + rank;
  okeys[pos] = key; ovals[pos] = val;
}

// ---------------- gather sorted rows ----------------
__global__ __launch_bounds__(256) void gather_rows(const float* __restrict__ src,
                                                   const unsigned* __restrict__ idx,
                                                   float* __restrict__ dst) {
  int g = blockIdx.x * 256 + threadIdx.x;    // over N*32 float4 slots
  int row = g >> 5, seg = g & 31;
  ((float4*)dst)[(size_t)row * 32 + seg] =
      ((const float4*)src)[(size_t)idx[row] * 32 + seg];
}

// ---------------- fused q,k,v,g projection (bf16 WMMA) ----------------
__global__ __launch_bounds__(256) void proj_qkvg(const float* __restrict__ xf,
                                                 const unsigned* __restrict__ wpack,
                                                 unsigned short* __restrict__ qo,
                                                 unsigned short* __restrict__ ko_,
                                                 unsigned short* __restrict__ vo,
                                                 unsigned short* __restrict__ go) {
  __shared__ unsigned short Xs[128][136];     // 272B row stride (bank-staggered)
  int tid = threadIdx.x;
  int rowbase = blockIdx.x * 128;
  for (int i = tid; i < 128 * 32; i += 256) {
    int r = i >> 5, sg = i & 31;
    float4 f = ((const float4*)xf)[((size_t)(rowbase + r)) * 32 + sg];
    unsigned* p = (unsigned*)&Xs[r][sg * 4];
    p[0] = packbf2(f.x, f.y); p[1] = packbf2(f.z, f.w);
  }
  __syncthreads();
  int w = tid >> 5, lane = tid & 31;
  Frag a[4];
#pragma unroll
  for (int kc = 0; kc < 4; ++kc) {          // A: 16x32 bf16 fragments
    const uint4* p = (const uint4*)&Xs[16 * w + (lane & 15)][kc * 32 + ((lane & 16) ? 16 : 0)];
    a[kc].q2[0] = p[0]; a[kc].q2[1] = p[1];
  }
  int m0 = rowbase + 16 * w + ((lane & 16) ? 8 : 0);
#pragma unroll
  for (int ws2 = 0; ws2 < 4; ++ws2) {
    unsigned short* out = (ws2 == 0) ? qo : (ws2 == 1) ? ko_ : (ws2 == 2) ? vo : go;
    const unsigned* wb = wpack + ws2 * 8192;
#pragma unroll
    for (int np = 0; np < 4; ++np) {         // two N-tiles per step, static dbl-buffer
      Frag bA[4], bB[4];
#pragma unroll
      for (int kc = 0; kc < 4; ++kc) {
        const uint4* pA = (const uint4*)&wb[((kc * 8 + 2 * np) * 32 + lane) * 8];
        const uint4* pB = (const uint4*)&wb[((kc * 8 + 2 * np + 1) * 32 + lane) * 8];
        bA[kc].q2[0] = pA[0]; bA[kc].q2[1] = pA[1];
        bB[kc].q2[0] = pB[0]; bB[kc].q2[1] = pB[1];
      }
      f32x8 accA = zero8(), accB = zero8();
#pragma unroll
      for (int kc = 0; kc < 4; ++kc) {
        accA = wmma_bf16(a[kc], bA[kc], accA);
        accB = wmma_bf16(a[kc], bB[kc], accB);
      }
      int nA = 2 * np * 16 + (lane & 15);
#pragma unroll
      for (int r = 0; r < 8; ++r) {
        float vA = accA[r], vB = accB[r];
        if (ws2 == 3) {
          vA = vA * __builtin_amdgcn_rcpf(1.0f + __expf(-vA));   // silu
          vB = vB * __builtin_amdgcn_rcpf(1.0f + __expf(-vB));
        }
        out[(size_t)(m0 + r) * C_DIM + nA] = f2bf(vA);
        out[(size_t)(m0 + r) * C_DIM + nA + 16] = f2bf(vB);
      }
    }
  }
}

// ---------------- per-chunk A_c = (k*kv_d)^T @ v (32x32, K=64) ----------------
__global__ __launch_bounds__(32) void chunk_kv(const unsigned short* __restrict__ kb,
                                               const unsigned short* __restrict__ vb,
                                               float* __restrict__ SA) {
  __shared__ unsigned short kT[32][72];
  __shared__ unsigned short vT[32][72];
  int bid = blockIdx.x;
  int c = bid & 63, h = (bid >> 6) & 3, g = bid >> 8;
  int lane = threadIdx.x;
  float gamma = 1.0f - exp2f(-5.0f - (float)h);
  float lg = __log2f(gamma);
  size_t rowbase = (size_t)g * GS_LEN + (size_t)c * T_CH;
#pragma unroll
  for (int rr = 0; rr < 2; ++rr) {
    int t = lane * 2 + rr;
    union { uint4 q[4]; unsigned u[16]; } kr, vr;
    const uint4* kp = (const uint4*)(kb + (rowbase + t) * C_DIM + h * DH_DIM);
    const uint4* vp = (const uint4*)(vb + (rowbase + t) * C_DIM + h * DH_DIM);
    kr.q[0] = kp[0]; kr.q[1] = kp[1]; kr.q[2] = kp[2]; kr.q[3] = kp[3];
    vr.q[0] = vp[0]; vr.q[1] = vp[1]; vr.q[2] = vp[2]; vr.q[3] = vp[3];
    float kvd = exp2f((float)(63 - t) * lg);
#pragma unroll
    for (int d = 0; d < 32; ++d) {
      unsigned kh = (d & 1) ? (kr.u[d >> 1] >> 16) : (kr.u[d >> 1] & 0xFFFFu);
      unsigned vh = (d & 1) ? (vr.u[d >> 1] >> 16) : (vr.u[d >> 1] & 0xFFFFu);
      kT[d][t] = f2bf(bf2f(kh) * kvd);
      vT[d][t] = (unsigned short)vh;
    }
  }
  __syncthreads();
  Frag bV[4];                                    // [nt*2+kc], resident across mt
#pragma unroll
  for (int nt = 0; nt < 2; ++nt)
#pragma unroll
    for (int kc = 0; kc < 2; ++kc) {
      const uint4* p =
          (const uint4*)&vT[nt * 16 + (lane & 15)][kc * 32 + ((lane & 16) ? 16 : 0)];
      bV[nt * 2 + kc].q2[0] = p[0]; bV[nt * 2 + kc].q2[1] = p[1];
    }
  float* dst = SA + (size_t)bid * 1024;
#pragma unroll
  for (int mt = 0; mt < 2; ++mt) {
    Frag aF[2];
    int m = mt * 16 + (lane & 15);
#pragma unroll
    for (int kc = 0; kc < 2; ++kc) {
      const uint4* p = (const uint4*)&kT[m][kc * 32 + ((lane & 16) ? 16 : 0)];
      aF[kc].q2[0] = p[0]; aF[kc].q2[1] = p[1];
    }
#pragma unroll
    for (int nt = 0; nt < 2; ++nt) {
      int e = nt * 16 + (lane & 15);
      f32x8 acc = zero8();
      acc = wmma_bf16(aF[0], bV[nt * 2 + 0], acc);
      acc = wmma_bf16(aF[1], bV[nt * 2 + 1], acc);
      int r0 = mt * 16 + ((lane & 16) ? 8 : 0);
#pragma unroll
      for (int r = 0; r < 8; ++r) dst[(r0 + r) * 32 + e] = acc[r];
    }
  }
}

// ---------------- sequential decay scan over chunks (in-place A_c -> S_c) ----
__global__ __launch_bounds__(256) void chunk_scan(float* __restrict__ SA) {
  int gh = blockIdx.x;                   // gh = g*4 + h
  int h = gh & 3;
  float gamma = 1.0f - exp2f(-5.0f - (float)h);
  float dk = exp2f(64.0f * __log2f(gamma));   // gamma^T
  int t = threadIdx.x;
  float s0 = 0.f, s1 = 0.f, s2 = 0.f, s3 = 0.f;
  size_t base = (size_t)gh * 64 * 1024;
  for (int c = 0; c < 64; ++c) {
    float* p = SA + base + (size_t)c * 1024;
    float a0 = p[t], a1 = p[t + 256], a2 = p[t + 512], a3 = p[t + 768];
    p[t] = s0; p[t + 256] = s1; p[t + 512] = s2; p[t + 768] = s3;
    s0 = dk * s0 + a0; s1 = dk * s1 + a1; s2 = dk * s2 + a2; s3 = dk * s3 + a3;
  }
}

// ---------------- per-chunk attention: y = (q@k^T . D)@v + diag(cross_d)(q@S_c)
__global__ __launch_bounds__(32) void chunk_attn(const unsigned short* __restrict__ qb,
                                                 const unsigned short* __restrict__ kb,
                                                 const unsigned short* __restrict__ vb,
                                                 const float* __restrict__ SA,
                                                 unsigned short* __restrict__ yb) {
  __shared__ unsigned short vT[32][72];
  __shared__ unsigned short ST[32][40];
  __shared__ unsigned short At[64][72];
  __shared__ float Dpow[65];                 // gamma^e, e = 0..64
  int bid = blockIdx.x;
  int c = bid & 63, h = (bid >> 6) & 3, g = bid >> 8;
  int lane = threadIdx.x;
  float gamma = 1.0f - exp2f(-5.0f - (float)h);
  float lg = __log2f(gamma);
  size_t rowbase = (size_t)g * GS_LEN + (size_t)c * T_CH;
  for (int i = lane; i < 65; i += 32) Dpow[i] = exp2f((float)i * lg);
  // resident A (q) and B (k) fragments straight from global (layout-contiguous)
  Frag aq[4], bk[4];
#pragma unroll
  for (int mt = 0; mt < 4; ++mt) {
    const uint4* p = (const uint4*)(qb + (rowbase + mt * 16 + (lane & 15)) * C_DIM +
                                    h * DH_DIM + ((lane & 16) ? 16 : 0));
    aq[mt].q2[0] = p[0]; aq[mt].q2[1] = p[1];
  }
#pragma unroll
  for (int nt = 0; nt < 4; ++nt) {
    const uint4* p = (const uint4*)(kb + (rowbase + nt * 16 + (lane & 15)) * C_DIM +
                                    h * DH_DIM + ((lane & 16) ? 16 : 0));
    bk[nt].q2[0] = p[0]; bk[nt].q2[1] = p[1];
  }
  // stage v^T
#pragma unroll
  for (int rr = 0; rr < 2; ++rr) {
    int t = lane * 2 + rr;
    union { uint4 q[4]; unsigned u[16]; } vr;
    const uint4* vp = (const uint4*)(vb + (rowbase + t) * C_DIM + h * DH_DIM);
    vr.q[0] = vp[0]; vr.q[1] = vp[1]; vr.q[2] = vp[2]; vr.q[3] = vp[3];
#pragma unroll
    for (int d = 0; d < 32; ++d)
      vT[d][t] = (unsigned short)((d & 1) ? (vr.u[d >> 1] >> 16) : (vr.u[d >> 1] & 0xFFFFu));
  }
  // stage S_c^T (bf16)
  {
    const float* S = SA + (size_t)bid * 1024;
    const float4* sr = (const float4*)(S + lane * 32);
#pragma unroll
    for (int eq = 0; eq < 8; ++eq) {
      float4 f = sr[eq];
      ST[eq * 4 + 0][lane] = f2bf(f.x); ST[eq * 4 + 1][lane] = f2bf(f.y);
      ST[eq * 4 + 2][lane] = f2bf(f.z); ST[eq * 4 + 3][lane] = f2bf(f.w);
    }
  }
  __syncthreads();
  // attn = (q @ k^T) .* D   ->  At (bf16, row-major)
#pragma unroll
  for (int mt = 0; mt < 4; ++mt) {
#pragma unroll
    for (int nt = 0; nt < 4; ++nt) {
      f32x8 acc = zero8();
      acc = wmma_bf16(aq[mt], bk[nt], acc);
      int tb = mt * 16 + ((lane & 16) ? 8 : 0);
      int sc = nt * 16 + (lane & 15);
#pragma unroll
      for (int r = 0; r < 8; ++r) {
        int e = (tb + r) - sc;
        float val = (e >= 0) ? acc[r] * Dpow[e] : 0.0f;
        At[tb + r][sc] = f2bf(val);
      }
    }
  }
  __syncthreads();
  // resident B fragments for phase 2
  Frag bv[4], bs[2];
#pragma unroll
  for (int nt = 0; nt < 2; ++nt) {
#pragma unroll
    for (int kc = 0; kc < 2; ++kc) {
      const uint4* p =
          (const uint4*)&vT[nt * 16 + (lane & 15)][kc * 32 + ((lane & 16) ? 16 : 0)];
      bv[nt * 2 + kc].q2[0] = p[0]; bv[nt * 2 + kc].q2[1] = p[1];
    }
    const uint4* p = (const uint4*)&ST[nt * 16 + (lane & 15)][(lane & 16) ? 16 : 0];
    bs[nt].q2[0] = p[0]; bs[nt].q2[1] = p[1];
  }
  // y = At @ v + diag(cross_d) * (q @ S_c)
#pragma unroll
  for (int mt = 0; mt < 4; ++mt) {
    Frag aA[2];
#pragma unroll
    for (int kc = 0; kc < 2; ++kc) {
      const uint4* p = (const uint4*)&At[mt * 16 + (lane & 15)][kc * 32 + ((lane & 16) ? 16 : 0)];
      aA[kc].q2[0] = p[0]; aA[kc].q2[1] = p[1];
    }
#pragma unroll
    for (int nt = 0; nt < 2; ++nt) {
      int e = nt * 16 + (lane & 15);
      f32x8 accI = zero8(), accC = zero8();
      accI = wmma_bf16(aA[0], bv[nt * 2 + 0], accI);
      accI = wmma_bf16(aA[1], bv[nt * 2 + 1], accI);
      accC = wmma_bf16(aq[mt], bs[nt], accC);
      int tb = mt * 16 + ((lane & 16) ? 8 : 0);
#pragma unroll
      for (int r = 0; r < 8; ++r) {
        float cd = Dpow[tb + r + 1];       // gamma^(t+1)
        yb[(rowbase + tb + r) * C_DIM + h * DH_DIM + e] = f2bf(accI[r] + accC[r] * cd);
      }
    }
  }
}

// ---------------- out = x + (y .* gate) @ Wo, scattered through idx ----------
__global__ __launch_bounds__(256) void out_proj(const float* __restrict__ xf,
                                                const unsigned short* __restrict__ yb,
                                                const unsigned short* __restrict__ gb,
                                                const unsigned* __restrict__ wpackO,
                                                const unsigned* __restrict__ idx,
                                                float* __restrict__ out) {
  __shared__ unsigned short Ys[128][136];
  int tid = threadIdx.x;
  int rowbase = blockIdx.x * 128;
  for (int i = tid; i < 128 * 64; i += 256) {
    int r = i >> 6, c2 = i & 63;
    unsigned yu = ((const unsigned*)yb)[((size_t)(rowbase + r)) * 64 + c2];
    unsigned gu = ((const unsigned*)gb)[((size_t)(rowbase + r)) * 64 + c2];
    float a0 = bf2f(yu & 0xFFFFu) * bf2f(gu & 0xFFFFu);
    float a1 = bf2f(yu >> 16) * bf2f(gu >> 16);
    *(unsigned*)&Ys[r][c2 * 2] = packbf2(a0, a1);
  }
  __syncthreads();
  int w = tid >> 5, lane = tid & 31;
  Frag a[4];
#pragma unroll
  for (int kc = 0; kc < 4; ++kc) {
    const uint4* p = (const uint4*)&Ys[16 * w + (lane & 15)][kc * 32 + ((lane & 16) ? 16 : 0)];
    a[kc].q2[0] = p[0]; a[kc].q2[1] = p[1];
  }
  int m0 = rowbase + 16 * w + ((lane & 16) ? 8 : 0);
#pragma unroll
  for (int np = 0; np < 4; ++np) {           // two N-tiles per step, static dbl-buffer
    Frag bA[4], bB[4];
#pragma unroll
    for (int kc = 0; kc < 4; ++kc) {
      const uint4* pA = (const uint4*)&wpackO[((kc * 8 + 2 * np) * 32 + lane) * 8];
      const uint4* pB = (const uint4*)&wpackO[((kc * 8 + 2 * np + 1) * 32 + lane) * 8];
      bA[kc].q2[0] = pA[0]; bA[kc].q2[1] = pA[1];
      bB[kc].q2[0] = pB[0]; bB[kc].q2[1] = pB[1];
    }
    f32x8 accA = zero8(), accB = zero8();
#pragma unroll
    for (int kc = 0; kc < 4; ++kc) {
      accA = wmma_bf16(a[kc], bA[kc], accA);
      accB = wmma_bf16(a[kc], bB[kc], accB);
    }
    int nA = 2 * np * 16 + (lane & 15);
#pragma unroll
    for (int r = 0; r < 8; ++r) {
      unsigned drow = idx[m0 + r];
      out[(size_t)drow * C_DIM + nA] = xf[(size_t)(m0 + r) * C_DIM + nA] + accA[r];
      out[(size_t)drow * C_DIM + nA + 16] = xf[(size_t)(m0 + r) * C_DIM + nA + 16] + accB[r];
    }
  }
}

// ---------------- host-side orchestration ----------------
extern "C" void kernel_launch(void* const* d_in, const int* in_sizes, int n_in,
                              void* d_out, int out_size, void* d_ws, size_t ws_size,
                              hipStream_t stream) {
  (void)in_sizes; (void)n_in; (void)out_size; (void)ws_size;
  const float* feats  = (const float*)d_in[0];
  const int*   coords = (const int*)d_in[1];
  const float* Wsrc[10] = { (const float*)d_in[2], (const float*)d_in[3],
                            (const float*)d_in[4], (const float*)d_in[5],
                            (const float*)d_in[6], (const float*)d_in[7],
                            (const float*)d_in[8], (const float*)d_in[9],
                            (const float*)d_in[10], (const float*)d_in[11] };
  float* out = (float*)d_out;

  char* ws = (char*)d_ws;
  size_t off = 0;
  auto take = [&](size_t bytes) -> char* {
    char* p = ws + off;
    off += (bytes + 255) & ~(size_t)255;
    return p;
  };
  unsigned* WPACK = (unsigned*)take(10 * 8192 * sizeof(unsigned)); // 10 packed 128x128 bf16
  unsigned* keysA = (unsigned*)take(N_TOT * 4);
  unsigned* idxA  = (unsigned*)take(N_TOT * 4);
  unsigned* keysB = (unsigned*)take(N_TOT * 4);
  unsigned* idxB  = (unsigned*)take(N_TOT * 4);
  unsigned* hist  = (unsigned*)take(256 * NB_SORT * 4);
  float*    xf    = (float*)take((size_t)N_TOT * C_DIM * 4);
  unsigned short* qbuf = (unsigned short*)take((size_t)N_TOT * C_DIM * 2);
  unsigned short* kbuf = (unsigned short*)take((size_t)N_TOT * C_DIM * 2);
  unsigned short* vbuf = (unsigned short*)take((size_t)N_TOT * C_DIM * 2);
  unsigned short* gbuf = (unsigned short*)take((size_t)N_TOT * C_DIM * 2);
  unsigned short* ybuf = (unsigned short*)take((size_t)N_TOT * C_DIM * 2);
  float*    SA    = (float*)take((size_t)NGRP * H_NUM * NCH * 1024 * 4);

  // pack weights: slots [pb*5 + {0:q,1:k,2:v,3:g,4:o}], q gets DH^-0.5 folded in
  const float qscale = 0.17677669529663687f;   // 32^-0.5
  for (int i = 0; i < 10; ++i) {
    int pb = i / 5, wi = i % 5;
    float sc = (wi == 0) ? qscale : 1.0f;
    pack_w<<<32, 256, 0, stream>>>(Wsrc[i], WPACK + (size_t)(pb * 5 + wi) * 8192, sc);
  }

  for (int pb = 0; pb < 2; ++pb) {
    // --- stable LSD radix argsort of the spatial keys ---
    compute_keys<<<NB_SORT, 256, 0, stream>>>(coords, keysA, idxA, pb);
    unsigned *ck = keysA, *cv = idxA, *ok = keysB, *ov = idxB;
    for (int pass = 0; pass < 4; ++pass) {
      int shift = 8 * pass;
      radix_hist<<<NB_SORT, 256, 0, stream>>>(ck, hist, shift);
      radix_scan<<<1, 256, 0, stream>>>(hist);
      radix_scatter<<<NB_SORT, 256, 0, stream>>>(ck, cv, ok, ov, hist, shift);
      unsigned* t;
      t = ck; ck = ok; ok = t;
      t = cv; cv = ov; ov = t;
    }
    // after 4 passes sorted indices live in idxA (cv == idxA)
    const float* src = (pb == 0) ? feats : (const float*)out;
    gather_rows<<<N_TOT * 32 / 256, 256, 0, stream>>>(src, cv, xf);
    proj_qkvg<<<N_TOT / 128, 256, 0, stream>>>(xf, WPACK + (size_t)pb * 5 * 8192,
                                               qbuf, kbuf, vbuf, gbuf);
    chunk_kv<<<NGRP * H_NUM * NCH, 32, 0, stream>>>(kbuf, vbuf, SA);
    chunk_scan<<<NGRP * H_NUM, 256, 0, stream>>>(SA);
    chunk_attn<<<NGRP * H_NUM * NCH, 32, 0, stream>>>(qbuf, kbuf, vbuf, SA, ybuf);
    out_proj<<<N_TOT / 128, 256, 0, stream>>>(xf, ybuf, gbuf,
                                              WPACK + (size_t)(pb * 5 + 4) * 8192, cv, out);
  }
}
